// Attention2D_3083786518758
// MI455X (gfx1250) — matrix operand order repaired and checked by reading, not compiled, not run on hardware
//
#include <hip/hip_runtime.h>
#include <hip/hip_bf16.h>
#include <stdint.h>

// ---------------------------------------------------------------------------
// Attention2D for MI455X (gfx1250, wave32, WMMA bf16, async-to-LDS pipeline)
// ---------------------------------------------------------------------------

typedef __attribute__((ext_vector_type(16))) __bf16 v16bf;
typedef __attribute__((ext_vector_type(8)))  float  v8f;

#define B_      16
#define C_      256
#define HID_    512
#define O3_     1536
#define NSP_    4096
#define HEADS_  8
#define DH_     64

#define LDS_STRIDE 40   // bf16 elems per padded row (80B -> conflict-free frags)

// ---- CDNA5 async global->LDS copy (ASYNCcnt), guarded so compile never breaks
#if defined(__has_builtin)
#  if __has_builtin(__builtin_amdgcn_global_load_async_to_lds_b128)
#    define HAS_ASYNC_LDS 1
#  endif
#endif
#ifndef HAS_ASYNC_LDS
#  define HAS_ASYNC_LDS 0
#endif

#if HAS_ASYNC_LDS
// Exact builtin param types per hipcc diagnostic: non-const int4, AS1 / AS3.
typedef __attribute__((ext_vector_type(4))) int v4i;
typedef __attribute__((address_space(1))) v4i* gas_v4i;
typedef __attribute__((address_space(3))) v4i* las_v4i;

static __device__ __forceinline__ void async_copy16(const void* g, void* l) {
    // Low 32 bits of a generic LDS pointer are the LDS byte offset (ISA §10.2).
    __builtin_amdgcn_global_load_async_to_lds_b128(
        (gas_v4i)(uintptr_t)g, (las_v4i)(uint32_t)(uintptr_t)l, 0, 0);
}
#  if __has_builtin(__builtin_amdgcn_s_wait_asynccnt)
#    define WAIT_ASYNC(n) __builtin_amdgcn_s_wait_asynccnt(n)
#  else
#    define WAIT_ASYNC(n) asm volatile("s_wait_asynccnt %0" :: "i"(n) : "memory")
#  endif
#endif

// Build a 16-element bf16 fragment from two 16-byte chunks.
static __device__ __forceinline__ v16bf frag2(const __bf16* p0, const __bf16* p1) {
    union { uint4 q[2]; v16bf v; } u;
    u.q[0] = *(const uint4*)p0;
    u.q[1] = *(const uint4*)p1;
    return u.v;
}

static __device__ __forceinline__ v8f wmma_bf16(v16bf a, v16bf b, v8f c) {
    return __builtin_amdgcn_wmma_f32_16x16x32_bf16(
        false, a, false, b, (short)0, c, false, false);
}

// ===========================================================================
// Kernel 1: QKV projection.  qkv[b] = W_qkv(1536x256,f32) * X[b](256x4096,f32)
// Block: 256 thr (8 waves), tile 128(M) x 128(N) x 32(K). bf16 output ws.
// ===========================================================================
__global__ __launch_bounds__(256)
void attn2d_qkv_gemm(const float* __restrict__ x,
                     const float* __restrict__ w_qkv,
                     __bf16* __restrict__ qkv) {
    __shared__ __align__(16) __bf16 lA [128 * LDS_STRIDE];
    __shared__ __align__(16) __bf16 lBT[128 * LDS_STRIDE];

    const int m_base = blockIdx.y * 128;
    const int n_base = blockIdx.x * 128;
    const int b      = blockIdx.z;

    const int tid  = threadIdx.x;
    const int lane = tid & 31;
    const int wv   = tid >> 5;
    const int lmod = lane & 15;
    const int lhi  = lane >> 4;
    const int mt0  = (wv >> 1) * 2;
    const int nt0  = (wv & 1) * 4;

    const int ar    = tid >> 1;
    const int ahalf = (tid & 1) * 16;
    const int kkr   = tid >> 3;
    const int cseg  = (tid & 7) * 16;

    v8f acc[2][4];
#pragma unroll
    for (int i = 0; i < 2; ++i)
#pragma unroll
        for (int j = 0; j < 4; ++j) acc[i][j] = (v8f)0.0f;

    const float* xb = x + (size_t)b * C_ * NSP_;

    for (int kt = 0; kt < C_; kt += 32) {
        {   // stage A (weights, f32 -> bf16, native v_cvt_pk_bf16_f32)
            const float* ap = w_qkv + (size_t)(m_base + ar) * C_ + kt + ahalf;
            __bf16* dst = &lA[ar * LDS_STRIDE + ahalf];
#pragma unroll
            for (int t = 0; t < 16; t += 4) {
                float4 f = *(const float4*)(ap + t);
                dst[t + 0] = (__bf16)f.x; dst[t + 1] = (__bf16)f.y;
                dst[t + 2] = (__bf16)f.z; dst[t + 3] = (__bf16)f.w;
            }
            if (kt + 32 < C_) __builtin_prefetch(ap + 32, 0, 3);
        }
        {   // stage B transposed (x, f32 -> bf16): lBT[n][k]
            const float* bp = xb + (size_t)(kt + kkr) * NSP_ + n_base + cseg;
#pragma unroll
            for (int t = 0; t < 16; t += 4) {
                float4 f = *(const float4*)(bp + t);
                lBT[(cseg + t + 0) * LDS_STRIDE + kkr] = (__bf16)f.x;
                lBT[(cseg + t + 1) * LDS_STRIDE + kkr] = (__bf16)f.y;
                lBT[(cseg + t + 2) * LDS_STRIDE + kkr] = (__bf16)f.z;
                lBT[(cseg + t + 3) * LDS_STRIDE + kkr] = (__bf16)f.w;
            }
            if (kt + 32 < C_) __builtin_prefetch(bp + (size_t)32 * NSP_, 0, 3);
        }
        __syncthreads();

        v16bf afr[2], bfr[4];
#pragma unroll
        for (int i = 0; i < 2; ++i) {
            const int base = ((mt0 + i) * 16 + lmod) * LDS_STRIDE + lhi * 8;
            afr[i] = frag2(&lA[base], &lA[base + 16]);
        }
#pragma unroll
        for (int j = 0; j < 4; ++j) {
            const int base = ((nt0 + j) * 16 + lmod) * LDS_STRIDE + lhi * 8;
            bfr[j] = frag2(&lBT[base], &lBT[base + 16]);
        }
#pragma unroll
        for (int i = 0; i < 2; ++i)
#pragma unroll
            for (int j = 0; j < 4; ++j)
                acc[i][j] = wmma_bf16(afr[i], bfr[j], acc[i][j]);
        __syncthreads();
    }

    __bf16* qb = qkv + (size_t)b * O3_ * NSP_;
#pragma unroll
    for (int i = 0; i < 2; ++i)
#pragma unroll
        for (int j = 0; j < 4; ++j) {
            const int n = n_base + (nt0 + j) * 16 + lmod;
#pragma unroll
            for (int vg = 0; vg < 8; ++vg) {
                const int m = m_base + (mt0 + i) * 16 + vg + lhi * 8;
                qb[(size_t)m * NSP_ + n] = (__bf16)acc[i][j][vg];
            }
        }
}

// ===========================================================================
// Kernel 2: attention per (b, head).  Block = 128 thr (4 waves).
// Phase 1 double-buffers q/k strips into LDS with async-to-LDS copies.
// ===========================================================================
__global__ __launch_bounds__(128)
void attn2d_attention(const __bf16* __restrict__ qkv,
                      __bf16* __restrict__ aout) {
    __shared__ __align__(16) float  simf [64 * 72];   // 18 KB
    __shared__ __align__(16) __bf16 attnb[64 * 72];   //  9 KB
    __shared__ __align__(16) __bf16 vt   [64 * 72];   //  9 KB
#if HAS_ASYNC_LDS
    __shared__ __align__(16) __bf16 qtile[2][64 * LDS_STRIDE];  // 2 x 5 KB
    __shared__ __align__(16) __bf16 ktile[2][64 * LDS_STRIDE];  // 2 x 5 KB
#endif

    const int h = blockIdx.x;
    const int b = blockIdx.y;
    const int tid  = threadIdx.x;
    const int lane = tid & 31;
    const int wv   = tid >> 5;           // 0..3 -> M strip
    const int lmod = lane & 15;
    const int lhi  = lane >> 4;

    const __bf16* q = qkv + ((size_t)b * O3_ +            h * DH_) * NSP_;
    const __bf16* k = qkv + ((size_t)b * O3_ + HID_     + h * DH_) * NSP_;
    const __bf16* v = qkv + ((size_t)b * O3_ + 2 * HID_ + h * DH_) * NSP_;

    // ---------------- phase 1: sim = q k^T over n=4096 ----------------
    v8f acc[4];
#pragma unroll
    for (int j = 0; j < 4; ++j) acc[j] = (v8f)0.0f;

#if HAS_ASYNC_LDS
    // copier mapping: thread -> (row 0..63, 2 x 16B chunks)
    const int cr  = tid >> 1;
    const int ce  = (tid & 1) * 16;      // element offset of this thread's 2 chunks
    auto issue_tile = [&](int buf, int k0) {
        const __bf16* qs = q + (size_t)cr * NSP_ + k0 + ce;
        __bf16*       qd = &qtile[buf][cr * LDS_STRIDE + ce];
        async_copy16(qs, qd);  async_copy16(qs + 8, qd + 8);
        const __bf16* ks = k + (size_t)cr * NSP_ + k0 + ce;
        __bf16*       kd = &ktile[buf][cr * LDS_STRIDE + ce];
        async_copy16(ks, kd);  async_copy16(ks + 8, kd + 8);
    };

    issue_tile(0, 0);
    for (int k0 = 0; k0 < NSP_; k0 += 32) {
        const int cur = (k0 >> 5) & 1;
        if (k0 + 32 < NSP_) { issue_tile(cur ^ 1, k0 + 32); WAIT_ASYNC(4); }
        else                { WAIT_ASYNC(0); }
        __syncthreads();

        const __bf16* qr = &qtile[cur][(wv * 16 + lmod) * LDS_STRIDE + lhi * 8];
        v16bf afr = frag2(qr, qr + 16);
#pragma unroll
        for (int j = 0; j < 4; ++j) {
            const __bf16* kr = &ktile[cur][(j * 16 + lmod) * LDS_STRIDE + lhi * 8];
            acc[j] = wmma_bf16(afr, frag2(kr, kr + 16), acc[j]);
        }
        __syncthreads();
    }
#else
    const __bf16* qr0 = q + (size_t)(wv * 16 + lmod) * NSP_ + lhi * 8;
#pragma unroll 2
    for (int k0 = 0; k0 < NSP_; k0 += 32) {
        v16bf afr = frag2(qr0 + k0, qr0 + k0 + 16);
#pragma unroll
        for (int j = 0; j < 4; ++j) {
            const __bf16* kr = k + (size_t)(j * 16 + lmod) * NSP_ + k0 + lhi * 8;
            acc[j] = wmma_bf16(afr, frag2(kr, kr + 16), acc[j]);
        }
    }
#endif

    // spill sim (scaled by dim_head^-0.5)
#pragma unroll
    for (int j = 0; j < 4; ++j)
#pragma unroll
        for (int vg = 0; vg < 8; ++vg)
            simf[(wv * 16 + vg + lhi * 8) * 72 + j * 16 + lmod] = acc[j][vg] * 0.125f;
    __syncthreads();

    // ---------------- softmax (one row per thread) ----------------
    if (tid < 64) {
        float* row = &simf[tid * 72];
        float m = row[0];
#pragma unroll 8
        for (int c = 1; c < 64; ++c) m = fmaxf(m, row[c]);
        float s = 0.0f;
#pragma unroll 8
        for (int c = 0; c < 64; ++c) { float e = __expf(row[c] - m); row[c] = e; s += e; }
        const float inv = __frcp_rn(s);
#pragma unroll 8
        for (int c = 0; c < 64; ++c) attnb[tid * 72 + c] = (__bf16)(row[c] * inv);
    }
    __syncthreads();

    // ---------------- phase 2: out = attn(64x64) * v(64x4096) ----------------
    const int kk   = tid >> 1;
    const int nseg = (tid & 1) * 32;
    __bf16* ab = aout + ((size_t)b * HID_ + h * DH_) * NSP_;

    for (int n0 = 0; n0 < NSP_; n0 += 64) {
        {   // stage v[64][n0:n0+64] transposed -> vt[n][k]
            const __bf16* vr = v + (size_t)kk * NSP_ + n0 + nseg;
            union { uint4 q[4]; __bf16 hsv[32]; } buf;
#pragma unroll
            for (int t = 0; t < 4; ++t) buf.q[t] = ((const uint4*)vr)[t];
#pragma unroll
            for (int t = 0; t < 32; ++t) vt[(nseg + t) * 72 + kk] = buf.hsv[t];
        }
        __syncthreads();

        v8f oacc[4];
#pragma unroll
        for (int j = 0; j < 4; ++j) oacc[j] = (v8f)0.0f;
#pragma unroll
        for (int ks = 0; ks < 64; ks += 32) {
            const __bf16* ar = &attnb[(wv * 16 + lmod) * 72 + ks + lhi * 8];
            v16bf afr = frag2(ar, ar + 16);
#pragma unroll
            for (int j = 0; j < 4; ++j) {
                const __bf16* br = &vt[(j * 16 + lmod) * 72 + ks + lhi * 8];
                oacc[j] = wmma_bf16(afr, frag2(br, br + 16), oacc[j]);
            }
        }
#pragma unroll
        for (int j = 0; j < 4; ++j)
#pragma unroll
            for (int vg = 0; vg < 8; ++vg) {
                const int m = wv * 16 + vg + lhi * 8;
                ab[(size_t)m * NSP_ + n0 + j * 16 + lmod] = (__bf16)oacc[j][vg];
            }
        __syncthreads();
    }
}

// ===========================================================================
// Kernel 3: out projection. y[b] = W_out(256x512,f32)*aout[b](512x4096,bf16)+bias
// ===========================================================================
__global__ __launch_bounds__(256)
void attn2d_out_gemm(const __bf16* __restrict__ aout,
                     const float* __restrict__ w_out,
                     const float* __restrict__ b_out,
                     float* __restrict__ y) {
    __shared__ __align__(16) __bf16 lA [128 * LDS_STRIDE];
    __shared__ __align__(16) __bf16 lBT[128 * LDS_STRIDE];

    const int m_base = blockIdx.y * 128;
    const int n_base = blockIdx.x * 128;
    const int b      = blockIdx.z;

    const int tid  = threadIdx.x;
    const int lane = tid & 31;
    const int wv   = tid >> 5;
    const int lmod = lane & 15;
    const int lhi  = lane >> 4;
    const int mt0  = (wv >> 1) * 2;
    const int nt0  = (wv & 1) * 4;

    const int ar    = tid >> 1;
    const int ahalf = (tid & 1) * 16;
    const int kkr   = tid >> 3;
    const int cseg  = (tid & 7) * 16;

    v8f acc[2][4];
#pragma unroll
    for (int i = 0; i < 2; ++i)
#pragma unroll
        for (int j = 0; j < 4; ++j) acc[i][j] = (v8f)0.0f;

    const __bf16* ab = aout + (size_t)b * HID_ * NSP_;

    for (int kt = 0; kt < HID_; kt += 32) {
        {   // stage A (w_out f32 -> bf16)
            const float* ap = w_out + (size_t)(m_base + ar) * HID_ + kt + ahalf;
            __bf16* dst = &lA[ar * LDS_STRIDE + ahalf];
#pragma unroll
            for (int t = 0; t < 16; t += 4) {
                float4 f = *(const float4*)(ap + t);
                dst[t + 0] = (__bf16)f.x; dst[t + 1] = (__bf16)f.y;
                dst[t + 2] = (__bf16)f.z; dst[t + 3] = (__bf16)f.w;
            }
            if (kt + 32 < HID_) __builtin_prefetch(ap + 32, 0, 3);
        }
        {   // stage B transposed (already bf16)
            const __bf16* bp = ab + (size_t)(kt + kkr) * NSP_ + n_base + cseg;
            union { uint4 q[2]; __bf16 hsv[16]; } buf;
            buf.q[0] = ((const uint4*)bp)[0];
            buf.q[1] = ((const uint4*)bp)[1];
#pragma unroll
            for (int t = 0; t < 16; ++t)
                lBT[(cseg + t) * LDS_STRIDE + kkr] = buf.hsv[t];
            if (kt + 32 < HID_) __builtin_prefetch(bp + (size_t)32 * NSP_, 0, 3);
        }
        __syncthreads();

        v16bf afr[2], bfr[4];
#pragma unroll
        for (int i = 0; i < 2; ++i) {
            const int base = ((mt0 + i) * 16 + lmod) * LDS_STRIDE + lhi * 8;
            afr[i] = frag2(&lA[base], &lA[base + 16]);
        }
#pragma unroll
        for (int j = 0; j < 4; ++j) {
            const int base = ((nt0 + j) * 16 + lmod) * LDS_STRIDE + lhi * 8;
            bfr[j] = frag2(&lBT[base], &lBT[base + 16]);
        }
#pragma unroll
        for (int i = 0; i < 2; ++i)
#pragma unroll
            for (int j = 0; j < 4; ++j)
                acc[i][j] = wmma_bf16(afr[i], bfr[j], acc[i][j]);
        __syncthreads();
    }

    float* yb = y + (size_t)b * C_ * NSP_;
#pragma unroll
    for (int i = 0; i < 2; ++i)
#pragma unroll
        for (int j = 0; j < 4; ++j) {
            const int n = n_base + (nt0 + j) * 16 + lmod;
#pragma unroll
            for (int vg = 0; vg < 8; ++vg) {
                const int m = m_base + (mt0 + i) * 16 + vg + lhi * 8;
                yb[(size_t)m * NSP_ + n] = acc[i][j][vg] + b_out[m];
            }
        }
}

// ===========================================================================
// Launch
// ===========================================================================
extern "C" void kernel_launch(void* const* d_in, const int* in_sizes, int n_in,
                              void* d_out, int out_size, void* d_ws, size_t ws_size,
                              hipStream_t stream) {
    const float* x     = (const float*)d_in[0];   // [16,256,64,64]
    const float* w_qkv = (const float*)d_in[1];   // [1536,256]
    const float* w_out = (const float*)d_in[2];   // [256,512]
    const float* b_out = (const float*)d_in[3];   // [256]
    float*       y     = (float*)d_out;           // [16,256,64,64]

    const size_t QKV_BYTES = (size_t)B_ * O3_ * NSP_ * sizeof(__bf16);   // 201.3 MB
    __bf16* qkv_ws  = (__bf16*)d_ws;
    __bf16* aout_ws = (__bf16*)((char*)d_ws + QKV_BYTES);                // +67.1 MB

    attn2d_qkv_gemm<<<dim3(NSP_ / 128, O3_ / 128, B_), 256, 0, stream>>>(
        x, w_qkv, qkv_ws);

    attn2d_attention<<<dim3(HEADS_, B_), 128, 0, stream>>>(qkv_ws, aout_ws);

    attn2d_out_gemm<<<dim3(NSP_ / 128, C_ / 128, B_), 256, 0, stream>>>(
        aout_ws, w_out, b_out, y);
}
